// TrajectoryDecoder_87016037417391
// MI455X (gfx1250) — matrix operand order repaired and checked by reading, not compiled, not run on hardware
//
#include <hip/hip_runtime.h>

// ---------------------------------------------------------------------------
// Trajectory decoder for MI455X (gfx1250, wave32, WMMA bf16 16x16x32).
// B=4096, F=256, H=256, M=3 modes, T derived from out_size (30).
//  * gx0_const = context @ w_ih0[:,2:]^T + b_ih0 precomputed once per mode
//    (context is time-invariant; per-step layer-0 gx is a rank-2 pos update).
//  * Main kernel: grid (3, 64), 512 threads = 16 waves, 64 batch rows/block.
//    Wave w owns column tile w inside each gate third (r=w, z=16+w, n=32+w),
//    so r/z/n gate math happens fully in-register on the WMMA C layout.
//  * Layer-1 r/z gates accumulate both h0n@w_ih1^T and h1@w_hh1^T into one
//    WMMA accumulator (bias-initialized); only the n-gate keeps xn/hn split.
//  * Weights streamed bf16 from L2 (3.5MB resident, 192 blocks sharing).
//  * Hidden states in LDS: fp32 mirrors + bank-conflict-padded bf16 mirrors.
// ---------------------------------------------------------------------------

#define B_   4096
#define F_   256
#define H_   256
#define G_   768     // 3*H
#define M_   3
#define MT_  64      // batch rows per block (main kernel)
#define NW_  16      // waves per block
#define HS_  264     // padded bf16 row stride (+8 halves = odd 16B-granule rotation)
#define MT3_ 32      // batch rows per block (precompute GEMM)

typedef __attribute__((ext_vector_type(16))) __bf16 v16bf;
typedef __attribute__((ext_vector_type(8)))  float  v8f;

union BV { v16bf v; uint4 u[2]; };

__device__ __forceinline__ unsigned short f2bf(float f) {
  unsigned u = __float_as_uint(f);
  u += 0x7FFFu + ((u >> 16) & 1u);          // round-to-nearest-even
  return (unsigned short)(u >> 16);
}
__device__ __forceinline__ float sigm(float x) { return 1.0f / (1.0f + __expf(-x)); }
__device__ __forceinline__ float tanh_f(float x) {
  float e = __expf(2.0f * x);
  return (e - 1.0f) / (e + 1.0f);
}

#define WMMA_BF16(A, Bm, C) \
  __builtin_amdgcn_wmma_f32_16x16x32_bf16(false, (A), false, (Bm), (short)0, (C), false, false)

// A operand (16x32 bf16, M x K) from LDS, padded row stride HS_:
// lane<16: row=lane,   K chunks at k*32+{0..7, 16..23}
// lane>=16:row=lane-16,K chunks at k*32+{8..15,24..31}
__device__ __forceinline__ v16bf loadA_lds(const unsigned short* hb, int mt, int k, int lane) {
  int row  = mt * 16 + (lane & 15);
  int koff = k * 32 + ((lane >> 4) << 3);
  BV a;
  a.u[0] = *(const uint4*)(hb + row * HS_ + koff);
  a.u[1] = *(const uint4*)(hb + row * HS_ + koff + 16);
  return a.v;
}
// Same A layout from global bf16 [B][256] (precompute GEMM), unpadded stride.
__device__ __forceinline__ v16bf loadA_g(const unsigned short* X, int base_row, int mt, int k, int lane) {
  int row  = base_row + mt * 16 + (lane & 15);
  int koff = k * 32 + ((lane >> 4) << 3);
  BV a;
  a.u[0] = *(const uint4*)(X + (size_t)row * H_ + koff);
  a.u[1] = *(const uint4*)(X + (size_t)row * H_ + koff + 16);
  return a.v;
}
// B operand (32x16 bf16, K x N) where B[k][n] = W[ct*16+n][k*32+k'].
// lane<16: col n=lane, K={0..7,16..23}; lane>=16: K={8..15,24..31}.
__device__ __forceinline__ v16bf loadB_g(const unsigned short* W, int ct, int k, int lane) {
  int row  = ct * 16 + (lane & 15);
  int koff = k * 32 + ((lane >> 4) << 3);
  BV b;
  const unsigned short* p = W + (size_t)row * H_ + koff;
  b.u[0] = *(const uint4*)(p);
  b.u[1] = *(const uint4*)(p + 16);
  return b.v;
}

// ---------------------------------------------------------------------------
__global__ __launch_bounds__(256) void f32_to_bf16_k(const float* __restrict__ src,
                                                     unsigned short* __restrict__ dst, int n) {
  for (int i = blockIdx.x * blockDim.x + threadIdx.x; i < n; i += gridDim.x * blockDim.x)
    dst[i] = f2bf(src[i]);
}

// Extract context columns [2:258) of w_ih0 (M,768,258) -> bf16 (M,768,256)
__global__ __launch_bounds__(256) void wih0_ctx_bf16_k(const float* __restrict__ w_ih0,
                                                       unsigned short* __restrict__ dst, int n) {
  for (int i = blockIdx.x * blockDim.x + threadIdx.x; i < n; i += gridDim.x * blockDim.x) {
    int ro = i >> 8;          // mode*768 + out_row
    int k  = i & 255;
    dst[i] = f2bf(w_ih0[(size_t)ro * 258 + 2 + k]);
  }
}

// ---------------------------------------------------------------------------
// Mode probabilities: relu(ctx @ w1^T + b1) @ w2^T + b2 -> softmax
__global__ __launch_bounds__(64) void mode_probs_k(const float* __restrict__ ctx,
                                                   const float* __restrict__ w1,
                                                   const float* __restrict__ b1,
                                                   const float* __restrict__ w2,
                                                   const float* __restrict__ b2,
                                                   float* __restrict__ outp) {
  int b = blockIdx.x, j = threadIdx.x;
  __shared__ float hbuf[64];
  __shared__ float lg[4];
  const float* cr = ctx + (size_t)b * F_;
  float s = b1[j];
  for (int k = 0; k < F_; ++k) s += cr[k] * w1[j * F_ + k];
  hbuf[j] = fmaxf(s, 0.f);
  __syncthreads();
  if (j < M_) {
    float t = b2[j];
    for (int k = 0; k < 64; ++k) t += hbuf[k] * w2[j * 64 + k];
    lg[j] = t;
  }
  __syncthreads();
  if (j == 0) {
    float m = fmaxf(lg[0], fmaxf(lg[1], lg[2]));
    float e0 = __expf(lg[0] - m), e1 = __expf(lg[1] - m), e2 = __expf(lg[2] - m);
    float inv = 1.f / (e0 + e1 + e2);
    outp[(size_t)b * M_ + 0] = e0 * inv;
    outp[(size_t)b * M_ + 1] = e1 * inv;
    outp[(size_t)b * M_ + 2] = e2 * inv;
  }
}

// ---------------------------------------------------------------------------
// gx0_const[mode][b][col] = b_ih0 + sum_k ctx[b][k] * w_ih0c[mode][col][k]
__global__ __launch_bounds__(256) void gx0_precompute(const unsigned short* __restrict__ ctxb,
                                                      const unsigned short* __restrict__ wb_ih0c,
                                                      const float* __restrict__ b_ih0,
                                                      float* __restrict__ gx0c) {
  const int mode = blockIdx.x;
  const int b0   = blockIdx.y * MT3_;
  const int tid  = threadIdx.x;
  const int lane = tid & 31;
  const int wv   = tid >> 5;
  const int cn   = lane & 15;
  const int ch   = lane >> 4;

  const unsigned short* W = wb_ih0c + (size_t)mode * G_ * H_;
  const float* bias = b_ih0 + mode * G_;

  v8f acc[2][6];
#pragma unroll
  for (int q = 0; q < 6; ++q) {
    float bv = bias[(wv + 8 * q) * 16 + cn];
#pragma unroll
    for (int mt = 0; mt < 2; ++mt)
#pragma unroll
      for (int i = 0; i < 8; ++i) acc[mt][q][i] = bv;
  }
#pragma unroll
  for (int k = 0; k < 8; ++k) {
    v16bf A0 = loadA_g(ctxb, b0, 0, k, lane);
    v16bf A1 = loadA_g(ctxb, b0, 1, k, lane);
#pragma unroll
    for (int q = 0; q < 6; ++q) {
      v16bf Bt = loadB_g(W, wv + 8 * q, k, lane);
      acc[0][q] = WMMA_BF16(A0, Bt, acc[0][q]);
      acc[1][q] = WMMA_BF16(A1, Bt, acc[1][q]);
    }
  }
#pragma unroll
  for (int q = 0; q < 6; ++q) {
    int col = (wv + 8 * q) * 16 + cn;
#pragma unroll
    for (int mt = 0; mt < 2; ++mt)
#pragma unroll
      for (int i = 0; i < 8; ++i) {
        int ml = mt * 16 + 8 * ch + i;
        gx0c[((size_t)mode * B_ + b0 + ml) * G_ + col] = acc[mt][q][i];
      }
  }
}

// ---------------------------------------------------------------------------
// Main recurrent decoder. Grid (M_, B_/MT_), 512 threads (16 waves).
__global__ __launch_bounds__(512) void traj_decode(
    const float* __restrict__ ctx, const float* __restrict__ gx0c,
    const unsigned short* __restrict__ wb_hh0, const unsigned short* __restrict__ wb_ih1,
    const unsigned short* __restrict__ wb_hh1,
    const float* __restrict__ b_hh0, const float* __restrict__ b_ih1,
    const float* __restrict__ b_hh1, const float* __restrict__ w_ih0,
    const float* __restrict__ out_w, const float* __restrict__ out_b,
    int T, float* __restrict__ out_traj) {
  const int mode = blockIdx.x;
  const int b0   = blockIdx.y * MT_;
  const int tid  = threadIdx.x;
  const int lane = tid & 31;
  const int wv   = tid >> 5;          // 0..15 : owns gate-column tile wv
  const int cn   = lane & 15;
  const int ch   = lane >> 4;

  __shared__ unsigned short h0b[MT_ * HS_];   // padded bf16 mirrors
  __shared__ unsigned short h1b[MT_ * HS_];
  __shared__ float h0f[MT_ * H_];             // fp32 state
  __shared__ float h1f[MT_ * H_];
  __shared__ float red[MT_ * 8 * 2];          // delta partial sums
  __shared__ float posL[MT_ * 2];
  __shared__ float cumL[MT_ * 2];
  __shared__ float owL[2 * H_];
  __shared__ float obL[2];

  for (int i = tid; i < MT_ * H_; i += 512) {
    int r = i >> 8, c = i & 255;
    float v = ctx[(size_t)(b0 + r) * F_ + c];
    h0f[i] = v; h1f[i] = v;
    unsigned short bv = f2bf(v);
    h0b[r * HS_ + c] = bv; h1b[r * HS_ + c] = bv;
  }
  if (tid < MT_ * 2) { posL[tid] = 0.f; cumL[tid] = 0.f; }
  if (tid < 2 * H_) owL[tid] = out_w[mode * 2 * H_ + tid];
  if (tid < 2) obL[tid] = out_b[mode * 2 + tid];
  __syncthreads();

  const unsigned short* Whh0 = wb_hh0 + (size_t)mode * G_ * H_;
  const unsigned short* Wih1 = wb_ih1 + (size_t)mode * G_ * H_;
  const unsigned short* Whh1 = wb_hh1 + (size_t)mode * G_ * H_;
  const float* Bhh0 = b_hh0 + mode * G_;
  const float* Bih1 = b_ih1 + mode * G_;
  const float* Bhh1 = b_hh1 + mode * G_;
  const float* Wih0 = w_ih0 + (size_t)mode * G_ * 258;
  const float* GX   = gx0c + (size_t)mode * B_ * G_;

  const int colr = wv * 16 + cn;       // this wave's r-column (also hidden col)
  const int colz = colr + 256;
  const int coln = colr + 512;

  for (int t = 0; t < T; ++t) {
    // =================== layer 0: gh0 via WMMA; gx0 folded into init =======
    v8f aR[4], aZ[4], aN[4];
    {
      float wr0 = Wih0[colr * 258 + 0], wr1 = Wih0[colr * 258 + 1];
      float wz0 = Wih0[colz * 258 + 0], wz1 = Wih0[colz * 258 + 1];
      float br = Bhh0[colr], bz = Bhh0[colz], bn = Bhh0[coln];
#pragma unroll
      for (int mt = 0; mt < 4; ++mt)
#pragma unroll
        for (int i = 0; i < 8; ++i) {
          int ml = mt * 16 + 8 * ch + i;
          float px = posL[ml * 2], py = posL[ml * 2 + 1];
          size_t ro = (size_t)(b0 + ml) * G_;
          aR[mt][i] = GX[ro + colr] + br + px * wr0 + py * wr1;
          aZ[mt][i] = GX[ro + colz] + bz + px * wz0 + py * wz1;
          aN[mt][i] = bn;                       // hn kept separate from xn
        }
    }
#pragma unroll
    for (int k = 0; k < 8; ++k) {
      v16bf Br = loadB_g(Whh0, wv,      k, lane);
      v16bf Bz = loadB_g(Whh0, wv + 16, k, lane);
      v16bf Bn = loadB_g(Whh0, wv + 32, k, lane);
#pragma unroll
      for (int mt = 0; mt < 4; ++mt) {
        v16bf A = loadA_lds(h0b, mt, k, lane);
        aR[mt] = WMMA_BF16(A, Br, aR[mt]);
        aZ[mt] = WMMA_BF16(A, Bz, aZ[mt]);
        aN[mt] = WMMA_BF16(A, Bn, aN[mt]);
      }
    }
    __syncthreads();  // all waves done reading h0b before anyone rewrites it
    {
      float wn0 = Wih0[coln * 258 + 0], wn1 = Wih0[coln * 258 + 1];
#pragma unroll
      for (int mt = 0; mt < 4; ++mt)
#pragma unroll
        for (int i = 0; i < 8; ++i) {
          int ml = mt * 16 + 8 * ch + i;
          float r = sigm(aR[mt][i]);
          float z = sigm(aZ[mt][i]);
          float px = posL[ml * 2], py = posL[ml * 2 + 1];
          float gxn = GX[(size_t)(b0 + ml) * G_ + coln] + px * wn0 + py * wn1;
          float n = tanh_f(gxn + r * aN[mt][i]);
          float ho = h0f[ml * H_ + colr];
          float hnew = (1.f - z) * n + z * ho;
          h0f[ml * H_ + colr] = hnew;
          h0b[ml * HS_ + colr] = f2bf(hnew);
        }
    }
    __syncthreads();

    // =================== layer 1: gx1 + gh1 (r/z fused into one acc) =======
    v8f cR[4], cZ[4], cXN[4], cHN[4];
    {
      float vr = Bih1[colr] + Bhh1[colr];
      float vz = Bih1[colz] + Bhh1[colz];
      float vx = Bih1[coln], vh = Bhh1[coln];
#pragma unroll
      for (int mt = 0; mt < 4; ++mt)
#pragma unroll
        for (int i = 0; i < 8; ++i) {
          cR[mt][i] = vr; cZ[mt][i] = vz;
          cXN[mt][i] = vx; cHN[mt][i] = vh;
        }
    }
#pragma unroll
    for (int k = 0; k < 8; ++k) {
      v16bf BiR = loadB_g(Wih1, wv,      k, lane);
      v16bf BhR = loadB_g(Whh1, wv,      k, lane);
      v16bf BiZ = loadB_g(Wih1, wv + 16, k, lane);
      v16bf BhZ = loadB_g(Whh1, wv + 16, k, lane);
      v16bf BiN = loadB_g(Wih1, wv + 32, k, lane);
      v16bf BhN = loadB_g(Whh1, wv + 32, k, lane);
#pragma unroll
      for (int mt = 0; mt < 4; ++mt) {
        v16bf Ax = loadA_lds(h0b, mt, k, lane);
        v16bf Ah = loadA_lds(h1b, mt, k, lane);
        cR[mt]  = WMMA_BF16(Ax, BiR, cR[mt]);
        cR[mt]  = WMMA_BF16(Ah, BhR, cR[mt]);
        cZ[mt]  = WMMA_BF16(Ax, BiZ, cZ[mt]);
        cZ[mt]  = WMMA_BF16(Ah, BhZ, cZ[mt]);
        cXN[mt] = WMMA_BF16(Ax, BiN, cXN[mt]);
        cHN[mt] = WMMA_BF16(Ah, BhN, cHN[mt]);
      }
    }
    __syncthreads();
#pragma unroll
    for (int mt = 0; mt < 4; ++mt)
#pragma unroll
      for (int i = 0; i < 8; ++i) {
        int ml = mt * 16 + 8 * ch + i;
        float r = sigm(cR[mt][i]);
        float z = sigm(cZ[mt][i]);
        float n = tanh_f(cXN[mt][i] + r * cHN[mt][i]);
        float ho = h1f[ml * H_ + colr];
        float hnew = (1.f - z) * n + z * ho;
        h1f[ml * H_ + colr] = hnew;
        h1b[ml * HS_ + colr] = f2bf(hnew);
      }
    __syncthreads();

    // ===== delta = h1 @ out_w^T + out_b; pos = delta; cum += delta ========
    {
      int row = tid >> 3, sub = tid & 7;          // 8 partials per row
      float s0 = 0.f, s1 = 0.f;
      const float* hr = h1f + row * H_ + sub * 32;
      const float* w0 = owL + sub * 32;
      const float* w1 = owL + H_ + sub * 32;
#pragma unroll
      for (int kk = 0; kk < 32; ++kk) {
        float v = hr[kk];
        s0 += v * w0[kk];
        s1 += v * w1[kk];
      }
      red[(row * 8 + sub) * 2 + 0] = s0;
      red[(row * 8 + sub) * 2 + 1] = s1;
    }
    __syncthreads();
    if (tid < MT_ * 2) {
      int row = tid >> 1, o = tid & 1;
      float s = obL[o];
#pragma unroll
      for (int q = 0; q < 8; ++q) s += red[(row * 8 + q) * 2 + o];
      float c = cumL[row * 2 + o] + s;
      cumL[row * 2 + o] = c;
      posL[row * 2 + o] = s;
      out_traj[((size_t)(b0 + row) * M_ + mode) * (size_t)(T * 2) + t * 2 + o] = c;
    }
    __syncthreads();
  }
}

// ---------------------------------------------------------------------------
extern "C" void kernel_launch(void* const* d_in, const int* in_sizes, int n_in,
                              void* d_out, int out_size, void* d_ws, size_t ws_size,
                              hipStream_t stream) {
  (void)in_sizes; (void)n_in; (void)ws_size;
  const float* ctx   = (const float*)d_in[0];
  const float* mp_w1 = (const float*)d_in[1];
  const float* mp_b1 = (const float*)d_in[2];
  const float* mp_w2 = (const float*)d_in[3];
  const float* mp_b2 = (const float*)d_in[4];
  const float* w_ih0 = (const float*)d_in[5];
  const float* w_hh0 = (const float*)d_in[6];
  const float* b_ih0 = (const float*)d_in[7];
  const float* b_hh0 = (const float*)d_in[8];
  const float* w_ih1 = (const float*)d_in[9];
  const float* w_hh1 = (const float*)d_in[10];
  const float* b_ih1 = (const float*)d_in[11];
  const float* b_hh1 = (const float*)d_in[12];
  const float* out_w = (const float*)d_in[13];
  const float* out_b = (const float*)d_in[14];

  // T derived from out_size: out = B*M*T*2 (traj) + B*M (probs)
  const int Th = (out_size - B_ * M_) / (B_ * M_ * 2);

  // workspace layout
  char* ws = (char*)d_ws;
  const size_t WH = (size_t)M_ * G_ * H_;             // 589824 elems per weight
  unsigned short* wb_hh0  = (unsigned short*)ws; ws += WH * 2;
  unsigned short* wb_ih1  = (unsigned short*)ws; ws += WH * 2;
  unsigned short* wb_hh1  = (unsigned short*)ws; ws += WH * 2;
  unsigned short* wb_ih0c = (unsigned short*)ws; ws += WH * 2;
  unsigned short* ctxb    = (unsigned short*)ws; ws += (size_t)B_ * F_ * 2;
  float* gx0c             = (float*)ws;          // M*B*G floats (~37.7 MB)

  float* out_traj  = (float*)d_out;
  float* out_probs = out_traj + (size_t)B_ * M_ * Th * 2;

  // 1) convert weights + context to bf16
  f32_to_bf16_k<<<512, 256, 0, stream>>>(w_hh0, wb_hh0, (int)WH);
  f32_to_bf16_k<<<512, 256, 0, stream>>>(w_ih1, wb_ih1, (int)WH);
  f32_to_bf16_k<<<512, 256, 0, stream>>>(w_hh1, wb_hh1, (int)WH);
  wih0_ctx_bf16_k<<<512, 256, 0, stream>>>(w_ih0, wb_ih0c, (int)WH);
  f32_to_bf16_k<<<512, 256, 0, stream>>>(ctx, ctxb, B_ * F_);

  // 2) precompute time-invariant layer-0 input gates (includes b_ih0)
  gx0_precompute<<<dim3(M_, B_ / MT3_), 256, 0, stream>>>(ctxb, wb_ih0c, b_ih0, gx0c);

  // 3) mode probabilities
  mode_probs_k<<<B_, 64, 0, stream>>>(ctx, mp_w1, mp_b1, mp_w2, mp_b2, out_probs);

  // 4) recurrent decode
  traj_decode<<<dim3(M_, B_ / MT_), 512, 0, stream>>>(
      ctx, gx0c, wb_hh0, wb_ih1, wb_hh1, b_hh0, b_ih1, b_hh1, w_ih0,
      out_w, out_b, Th, out_traj);
}
